// CRF_5995774345317
// MI455X (gfx1250) — matrix operand order repaired and checked by reading, not compiled, not run on hardware
//
#include <hip/hip_runtime.h>
#include <math.h>

#define HH   320
#define WWID 320
#define NPIX (HH*WWID)
#define NC   21

typedef __attribute__((ext_vector_type(16))) _Float16 v16h;
typedef __attribute__((ext_vector_type(8)))  float    v8f;

// ---------------------------------------------------------------- hashing
static __device__ __forceinline__ uint32_t hash_key(const int* k, int d) {
    uint32_t h = 2166136261u;
    for (int i = 0; i < d; ++i) { h ^= (uint32_t)k[i]; h *= 16777619u; }
    h ^= h >> 15; h *= 2654435761u; h ^= h >> 13;
    return h;
}

// ------------------------------------------------- lattice build (per pixel)
// MODE 0: bilateral (x/80, y/80, rgb/13), D=5.  MODE 1: spatial (x/3, y/3), D=2.
template<int D, int MODE>
__global__ void k_lattice(const float* __restrict__ img,
                          float* __restrict__ wsA, int* __restrict__ keysA) {
    int n = blockIdx.x * blockDim.x + threadIdx.x;
    if (n >= NPIX) return;
    const int L = D + 1;
    int row = n / WWID, col = n % WWID;
    float f[8];
    if (MODE == 0) {
        f[0] = col * (1.0f/80.0f); f[1] = row * (1.0f/80.0f);
        f[2] = img[n*3+0] * (1.0f/13.0f);
        f[3] = img[n*3+1] * (1.0f/13.0f);
        f[4] = img[n*3+2] * (1.0f/13.0f);
    } else {
        f[0] = col * (1.0f/3.0f); f[1] = row * (1.0f/3.0f);
    }
    float cf[D];
    #pragma unroll
    for (int i = 0; i < D; ++i) {
        float sc = sqrtf(2.0f/3.0f) * (float)(D+1) / sqrtf((float)((i+1)*(i+2)));
        cf[i] = f[i] * sc;
    }
    float csum[D+1]; csum[D] = 0.0f;
    for (int i = D-1; i >= 0; --i) csum[i] = csum[i+1] + cf[i];
    float el[D+1];
    el[0] = csum[0];
    #pragma unroll
    for (int i = 1; i <= D; ++i) el[i] = csum[i] - (float)i * cf[i-1];
    const float down = 1.0f / (float)(D+1);
    float rem0[D+1]; float rdsum = 0.0f;
    #pragma unroll
    for (int i = 0; i <= D; ++i) {
        float rd = rintf(el[i] * down);
        rem0[i] = rd * (float)(D+1);
        rdsum += rd;
    }
    int ssum = (int)rintf(rdsum);
    float diff[D+1];
    #pragma unroll
    for (int i = 0; i <= D; ++i) diff[i] = el[i] - rem0[i];
    int rank[D+1];
    #pragma unroll
    for (int i = 0; i <= D; ++i) {
        int r = 0;
        for (int j = 0; j <= D; ++j)
            r += ((diff[j] > diff[i]) || (diff[j] == diff[i] && j < i)) ? 1 : 0;
        rank[i] = r + ssum;
    }
    #pragma unroll
    for (int i = 0; i <= D; ++i) {
        if (rank[i] < 0)      { rank[i] += L; rem0[i] += (float)L; }
        else if (rank[i] > D) { rank[i] -= L; rem0[i] -= (float)L; }
    }
    float b[D+2];
    #pragma unroll
    for (int i = 0; i < D+2; ++i) b[i] = 0.0f;
    #pragma unroll
    for (int i = 0; i <= D; ++i) {
        float v = (el[i] - rem0[i]) * down;
        b[D - rank[i]]     += v;
        b[D + 1 - rank[i]] -= v;
    }
    b[0] += 1.0f + b[D+1];
    int key0[D];
    #pragma unroll
    for (int i = 0; i < D; ++i) key0[i] = (int)rintf(rem0[i]);
    #pragma unroll
    for (int r = 0; r <= D; ++r) {
        wsA[(size_t)n*L + r] = b[r];
        int* kp = keysA + ((size_t)n*L + r) * D;
        for (int i = 0; i < D; ++i)
            kp[i] = key0[i] + ((rank[i] < L - r) ? r : r - L);
    }
}

// ------------------------------------------------- hash insert / assign / os
template<int D>
__global__ void k_insert(const int* __restrict__ keysA, int* entry,
                         int* __restrict__ slotOf, int total, int capMask) {
    int t = blockIdx.x * blockDim.x + threadIdx.x;
    if (t >= total) return;
    const int* mk = keysA + (size_t)t * D;
    uint32_t h = hash_key(mk, D) & (uint32_t)capMask;
    for (;;) {
        int prev = atomicCAS(&entry[h], -1, t);
        if (prev == -1) { slotOf[t] = (int)h; return; }
        const int* pk = keysA + (size_t)prev * D;
        bool eq = true;
        for (int i = 0; i < D; ++i) eq &= (pk[i] == mk[i]);
        if (eq) { slotOf[t] = (int)h; return; }
        h = (h + 1) & (uint32_t)capMask;
    }
}

__global__ void k_assign(const int* __restrict__ entry, int* __restrict__ idArr,
                         int* counter, int cap) {
    int s = blockIdx.x * blockDim.x + threadIdx.x;
    if (s >= cap) return;
    if (entry[s] >= 0) idArr[s] = atomicAdd(counter, 1);
}

__global__ void k_os(const int* __restrict__ slotOf, const int* __restrict__ idArr,
                     int* __restrict__ osA, int total) {
    int t = blockIdx.x * blockDim.x + threadIdx.x;
    if (t >= total) return;
    osA[t] = idArr[slotOf[t]];
}

// ------------------------------------------------- neighbor tables for blur
template<int D>
static __device__ __forceinline__ int hlookup(const int* nk, const int* entry,
                                              const int* idArr, const int* keysA,
                                              int capMask) {
    uint32_t h = hash_key(nk, D) & (uint32_t)capMask;
    for (;;) {
        int e = entry[h];
        if (e < 0) return -1;
        const int* pk = keysA + (size_t)e * D;
        bool eq = true;
        for (int i = 0; i < D; ++i) eq &= (pk[i] == nk[i]);
        if (eq) return idArr[h];
        h = (h + 1) & (uint32_t)capMask;
    }
}

template<int D>
__global__ void k_neigh(const int* __restrict__ entry, const int* __restrict__ idArr,
                        const int* __restrict__ keysA, int* __restrict__ n1,
                        int* __restrict__ n2, int cap, int capMask, int Rmax) {
    int s = blockIdx.x * blockDim.x + threadIdx.x;
    if (s >= cap) return;
    int t = entry[s];
    if (t < 0) return;
    int myid = idArr[s];
    const int* mk = keysA + (size_t)t * D;
    int key[D];
    #pragma unroll
    for (int i = 0; i < D; ++i) key[i] = mk[i];
    for (int j = 0; j <= D; ++j) {
        int nk[D];
        for (int i = 0; i < D; ++i) nk[i] = key[i] - 1 + ((i == j) ? (D+1) : 0);
        n1[(size_t)j * Rmax + myid] = hlookup<D>(nk, entry, idArr, keysA, capMask);
        for (int i = 0; i < D; ++i) nk[i] = key[i] + 1 - ((i == j) ? (D+1) : 0);
        n2[(size_t)j * Rmax + myid] = hlookup<D>(nk, entry, idArr, keysA, capMask);
    }
}

// ------------------------------------------------- filter primitives
__global__ void k_clear(float* __restrict__ buf, const int* Mptr, int Cc) {
    long long M = *Mptr;
    long long idx = (long long)blockIdx.x * blockDim.x + threadIdx.x;
    if (idx < M * Cc) buf[idx] = 0.0f;
}

__global__ void k_splat(const float* __restrict__ val, const float* __restrict__ wsA,
                        const int* __restrict__ osA, float* __restrict__ buf,
                        int total, int Cc, int L) {
    int t = blockIdx.x * blockDim.x + threadIdx.x;
    if (t >= total) return;
    int n = t / L;
    __builtin_prefetch(&val[(size_t)n * Cc], 0, 0);   // global_prefetch_b8
    float w = wsA[t];
    int rowi = osA[t];
    for (int c = 0; c < Cc; ++c)
        atomicAdd(&buf[(size_t)rowi * Cc + c], w * val[(size_t)n * Cc + c]);
}

__global__ void k_splat_ones(const float* __restrict__ wsA, const int* __restrict__ osA,
                             float* __restrict__ buf, int total) {
    int t = blockIdx.x * blockDim.x + threadIdx.x;
    if (t >= total) return;
    atomicAdd(&buf[osA[t]], wsA[t]);
}

__global__ void k_blur(const float* __restrict__ in, float* __restrict__ out,
                       const int* __restrict__ n1, const int* __restrict__ n2,
                       const int* Mptr, int Cc) {
    long long M = *Mptr;
    long long idx = (long long)blockIdx.x * blockDim.x + threadIdx.x;
    if (idx >= M * Cc) return;
    int v = (int)(idx / Cc), c = (int)(idx % Cc);
    int i1 = n1[v], i2 = n2[v];
    float a  = in[idx];
    float x1 = (i1 >= 0) ? in[(size_t)i1 * Cc + c] : 0.0f;
    float x2 = (i2 >= 0) ? in[(size_t)i2 * Cc + c] : 0.0f;
    out[idx] = a + 0.5f * (x1 + x2);
}

template<int L>
__global__ void k_slice_norm(const float* __restrict__ wsA, const int* __restrict__ osA,
                             const float* __restrict__ buf, float* __restrict__ norm,
                             float alpha) {
    int n = blockIdx.x * blockDim.x + threadIdx.x;
    if (n >= NPIX) return;
    float s = 0.0f;
    #pragma unroll
    for (int j = 0; j < L; ++j)
        s += wsA[(size_t)n*L + j] * buf[osA[(size_t)n*L + j]];
    norm[n] = alpha * s + 1e-20f;
}

// ------------------------------------------------- WMMA slice:
// per 16-pixel tile: out(16x21) = A(16xKPAD) x B(KPADx21),
// A block-sparse barycentric weights, B gathered lattice rows, f16 in / f32 acc.
// B is staged TRANSPOSED ([channel][k]) so each lane's 16 fragment halves are
// 32 contiguous, 16B-aligned bytes -> ds_load_b128 instead of 16x ds_load_u16.
template<int L, int KPAD>
__global__ void __launch_bounds__(32)
k_slice_wmma(const float* __restrict__ wsA, const int* __restrict__ osA,
             const float* __restrict__ buf, float* __restrict__ outv, float alpha) {
    __shared__ __align__(16) _Float16 Ash[16][KPAD];
    __shared__ __align__(16) _Float16 BshT[32][KPAD];
    const int lane  = threadIdx.x;
    const int pbase = blockIdx.x * 16;

    // zero A (padding columns must be 0)
    _Float16* Af = &Ash[0][0];
    for (int i = lane; i < 16 * KPAD; i += 32) Af[i] = (_Float16)0.0f;
    __syncthreads();
    // A[m][m*L+j] = ws[m][j]
    if (lane < 16) {
        for (int j = 0; j < L; ++j)
            Ash[lane][lane*L + j] = (_Float16)wsA[(size_t)(pbase + lane)*L + j];
    }
    // BshT[c][k] = buf[os[m][j]][c], k = m*L+j ; zero-pad c>=21 and k >= 16L
    for (int k = lane; k < KPAD; k += 32) {
        if (k < 16 * L) {
            int m = k / L, j = k % L;
            int rowi = osA[(size_t)(pbase + m)*L + j];
            const float* br = buf + (size_t)rowi * NC;
            for (int c = 0;  c < NC; ++c) BshT[c][k] = (_Float16)br[c];
            for (int c = NC; c < 32; ++c) BshT[c][k] = (_Float16)0.0f;
        } else {
            for (int c = 0; c < 32; ++c) BshT[c][k] = (_Float16)0.0f;
        }
    }
    __syncthreads();

    const int m16 = lane & 15, hi = lane >> 4;
    #pragma unroll
    for (int ct = 0; ct < 2; ++ct) {
        v8f acc = {};
        #pragma unroll
        for (int ch = 0; ch < KPAD / 32; ++ch) {
            const int kc = ch * 32;
            v16h a, b;
            // A 16x32 f16 layout: lane = M row (mod 16); element j <-> K = hi*16 + j
            // (contiguous, matching B).
            const _Float16* ap = &Ash[m16][kc];
            #pragma unroll
            for (int i = 0; i < 8; ++i) {
                int kp = hi*16 + 2*i;
                a[2*i]   = ap[kp];
                a[2*i+1] = ap[kp + 1];
            }
            // B 32x16 f16 layout: lanes 0-15 hold K=0..15, lanes 16-31 K=16..31,
            // column = lane&15; element j <-> K = kc + hi*16 + j (contiguous).
            const _Float16* bp = &BshT[ct*16 + m16][kc + hi*16];
            #pragma unroll
            for (int j = 0; j < 16; ++j) b[j] = bp[j];
            acc = __builtin_amdgcn_wmma_f32_16x16x32_f16(
                      false, a, false, b, (short)0, acc, false, false);
        }
        #pragma unroll
        for (int v = 0; v < 8; ++v) {          // D layout: M = v + 8*hi, N = lane&15
            int m = v + 8*hi;
            int c = ct*16 + m16;
            if (c < NC) outv[(size_t)(pbase + m)*NC + c] = alpha * acc[v];
        }
    }
}

// ------------------------------------------------- softmax / combine
__global__ void k_softmax_init(const float* __restrict__ U, float* __restrict__ Q) {
    int n = blockIdx.x * blockDim.x + threadIdx.x;
    if (n >= NPIX) return;
    float l[NC]; float mx = -3.4e38f;
    for (int c = 0; c < NC; ++c) { l[c] = -U[(size_t)n*NC + c]; mx = fmaxf(mx, l[c]); }
    float s = 0.0f;
    for (int c = 0; c < NC; ++c) { l[c] = expf(l[c] - mx); s += l[c]; }
    float inv = 1.0f / s;
    for (int c = 0; c < NC; ++c) Q[(size_t)n*NC + c] = l[c] * inv;
}

__global__ void k_combine(const float* __restrict__ U, const float* __restrict__ tb,
                          const float* __restrict__ ts, const float* __restrict__ nb,
                          const float* __restrict__ ns, float* __restrict__ Q) {
    int n = blockIdx.x * blockDim.x + threadIdx.x;
    if (n >= NPIX) return;
    float inb = 10.0f / nb[n];
    float ins = 3.0f  / ns[n];
    float l[NC]; float mx = -3.4e38f;
    for (int c = 0; c < NC; ++c) {
        size_t i = (size_t)n*NC + c;
        l[c] = -U[i] + tb[i]*inb + ts[i]*ins;
        mx = fmaxf(mx, l[c]);
    }
    float s = 0.0f;
    for (int c = 0; c < NC; ++c) { l[c] = expf(l[c] - mx); s += l[c]; }
    float inv = 1.0f / s;
    for (int c = 0; c < NC; ++c) Q[(size_t)n*NC + c] = l[c] * inv;
}

// ------------------------------------------------- host orchestration
extern "C" void kernel_launch(void* const* d_in, const int* in_sizes, int n_in,
                              void* d_out, int out_size, void* d_ws, size_t ws_size,
                              hipStream_t stream) {
    (void)in_sizes; (void)n_in; (void)out_size; (void)ws_size;
    const float* U   = (const float*)d_in[0];   // (320,320,21)
    const float* img = (const float*)d_in[1];   // (320,320,3)
    float* outQ = (float*)d_out;

    const int DB = 5, LB = 6; const int RB = NPIX * LB; const int CAPB = 1 << 20;
    const int DS = 2, LS = 3; const int RS = NPIX * LS; const int CAPS = 1 << 19;
    (void)DB; (void)DS;

    size_t off = 0;
    auto alloc = [&](size_t bytes) -> void* {
        void* p = (char*)d_ws + off;
        off += (bytes + 255) & ~(size_t)255;
        return p;
    };
    // bilateral lattice
    float* wsB   = (float*)alloc((size_t)RB * 4);
    int*   osB   = (int*)  alloc((size_t)RB * 4);
    int*   keysB = (int*)  alloc((size_t)RB * 5 * 4);
    int*   entB  = (int*)  alloc((size_t)CAPB * 4);
    int*   idB   = (int*)  alloc((size_t)CAPB * 4);
    int*   slotB = (int*)  alloc((size_t)RB * 4);
    int*   n1B   = (int*)  alloc((size_t)LB * RB * 4);
    int*   n2B   = (int*)  alloc((size_t)LB * RB * 4);
    float* bufB0 = (float*)alloc((size_t)RB * NC * 4);
    float* bufB1 = (float*)alloc((size_t)RB * NC * 4);
    // spatial lattice
    float* wsS   = (float*)alloc((size_t)RS * 4);
    int*   osS   = (int*)  alloc((size_t)RS * 4);
    int*   keysS = (int*)  alloc((size_t)RS * 2 * 4);
    int*   entS  = (int*)  alloc((size_t)CAPS * 4);
    int*   idS   = (int*)  alloc((size_t)CAPS * 4);
    int*   slotS = (int*)  alloc((size_t)RS * 4);
    int*   n1S   = (int*)  alloc((size_t)LS * RS * 4);
    int*   n2S   = (int*)  alloc((size_t)LS * RS * 4);
    float* bufS0 = (float*)alloc((size_t)RS * NC * 4);
    float* bufS1 = (float*)alloc((size_t)RS * NC * 4);
    // shared
    float* normB = (float*)alloc((size_t)NPIX * 4);
    float* normS = (float*)alloc((size_t)NPIX * 4);
    float* Q     = (float*)alloc((size_t)NPIX * NC * 4);
    float* tmpB  = (float*)alloc((size_t)NPIX * NC * 4);
    float* tmpS  = (float*)alloc((size_t)NPIX * NC * 4);
    int*   cntB  = (int*)  alloc(4);
    int*   cntS  = (int*)  alloc(4);

    auto g = [](long long n) { return dim3((unsigned)((n + 255) / 256)); };
    const dim3 B256(256);

    hipMemsetAsync(entB, 0xFF, (size_t)CAPB * 4, stream);
    hipMemsetAsync(entS, 0xFF, (size_t)CAPS * 4, stream);
    hipMemsetAsync(cntB, 0, 4, stream);
    hipMemsetAsync(cntS, 0, 4, stream);

    // ---- build lattices
    k_lattice<5,0><<<g(NPIX), B256, 0, stream>>>(img, wsB, keysB);
    k_insert<5>   <<<g(RB),   B256, 0, stream>>>(keysB, entB, slotB, RB, CAPB - 1);
    k_assign      <<<g(CAPB), B256, 0, stream>>>(entB, idB, cntB, CAPB);
    k_os          <<<g(RB),   B256, 0, stream>>>(slotB, idB, osB, RB);
    k_neigh<5>    <<<g(CAPB), B256, 0, stream>>>(entB, idB, keysB, n1B, n2B, CAPB, CAPB - 1, RB);

    k_lattice<2,1><<<g(NPIX), B256, 0, stream>>>(img, wsS, keysS);
    k_insert<2>   <<<g(RS),   B256, 0, stream>>>(keysS, entS, slotS, RS, CAPS - 1);
    k_assign      <<<g(CAPS), B256, 0, stream>>>(entS, idS, cntS, CAPS);
    k_os          <<<g(RS),   B256, 0, stream>>>(slotS, idS, osS, RS);
    k_neigh<2>    <<<g(CAPS), B256, 0, stream>>>(entS, idS, keysS, n1S, n2S, CAPS, CAPS - 1, RS);

    const float alphaB = 32.0f / 33.0f;   // 1/(1+2^-5)
    const float alphaS = 4.0f / 5.0f;     // 1/(1+2^-2)

    // ---- normalization filters (C=1)
    {
        k_clear<<<g(RB), B256, 0, stream>>>(bufB0, cntB, 1);
        k_splat_ones<<<g(RB), B256, 0, stream>>>(wsB, osB, bufB0, RB);
        float* pin = bufB0; float* pout = bufB1;
        for (int j = 0; j <= 5; ++j) {
            k_blur<<<g(RB), B256, 0, stream>>>(pin, pout, n1B + (size_t)j*RB, n2B + (size_t)j*RB, cntB, 1);
            float* t = pin; pin = pout; pout = t;
        }
        k_slice_norm<6><<<g(NPIX), B256, 0, stream>>>(wsB, osB, pin, normB, alphaB);
    }
    {
        k_clear<<<g(RS), B256, 0, stream>>>(bufS0, cntS, 1);
        k_splat_ones<<<g(RS), B256, 0, stream>>>(wsS, osS, bufS0, RS);
        float* pin = bufS0; float* pout = bufS1;
        for (int j = 0; j <= 2; ++j) {
            k_blur<<<g(RS), B256, 0, stream>>>(pin, pout, n1S + (size_t)j*RS, n2S + (size_t)j*RS, cntS, 1);
            float* t = pin; pin = pout; pout = t;
        }
        k_slice_norm<3><<<g(NPIX), B256, 0, stream>>>(wsS, osS, pin, normS, alphaS);
    }

    // ---- mean-field iterations
    k_softmax_init<<<g(NPIX), B256, 0, stream>>>(U, Q);
    for (int it = 0; it < 5; ++it) {
        // bilateral filter(Q)
        k_clear<<<g((long long)RB * NC), B256, 0, stream>>>(bufB0, cntB, NC);
        k_splat<<<g(RB), B256, 0, stream>>>(Q, wsB, osB, bufB0, RB, NC, LB);
        float* pin = bufB0; float* pout = bufB1;
        for (int j = 0; j <= 5; ++j) {
            k_blur<<<g((long long)RB * NC), B256, 0, stream>>>(pin, pout, n1B + (size_t)j*RB, n2B + (size_t)j*RB, cntB, NC);
            float* t = pin; pin = pout; pout = t;
        }
        k_slice_wmma<6, 96><<<dim3(NPIX/16), dim3(32), 0, stream>>>(wsB, osB, pin, tmpB, alphaB);

        // spatial filter(Q)
        k_clear<<<g((long long)RS * NC), B256, 0, stream>>>(bufS0, cntS, NC);
        k_splat<<<g(RS), B256, 0, stream>>>(Q, wsS, osS, bufS0, RS, NC, LS);
        pin = bufS0; pout = bufS1;
        for (int j = 0; j <= 2; ++j) {
            k_blur<<<g((long long)RS * NC), B256, 0, stream>>>(pin, pout, n1S + (size_t)j*RS, n2S + (size_t)j*RS, cntS, NC);
            float* t = pin; pin = pout; pout = t;
        }
        k_slice_wmma<3, 64><<<dim3(NPIX/16), dim3(32), 0, stream>>>(wsS, osS, pin, tmpS, alphaS);

        k_combine<<<g(NPIX), B256, 0, stream>>>(U, tmpB, tmpS, normB, normS, Q);
    }

    hipMemcpyAsync(outQ, Q, (size_t)NPIX * NC * 4, hipMemcpyDeviceToDevice, stream);
}
